// ECGRNN_10505490006664
// MI455X (gfx1250) — matrix operand order repaired and checked by reading, not compiled, not run on hardware
//
#include <hip/hip_runtime.h>
#include <hip/hip_bf16.h>

typedef __attribute__((ext_vector_type(16))) _Float16 v16h;
typedef __attribute__((ext_vector_type(8)))  _Float16 v8h;
typedef __attribute__((ext_vector_type(4)))  _Float16 v4h;
typedef __attribute__((ext_vector_type(8)))  float    v8f;

#define B_   256
#define L_   1024
#define H_   128
#define FC1_ 64
#define K_TOT (L_ * H_)   // 131072

static __device__ __forceinline__ v8f wmma_f16(v16h a, v16h b, v8f c) {
    // D = A(16x32 f16) x B(32x16 f16) + C(16x16 f32)
    return __builtin_amdgcn_wmma_f32_16x16x32_f16(
        /*neg_a=*/false, a, /*neg_b=*/false, b,
        /*c_mod=*/(short)0, c, /*reuse_a=*/false, /*reuse_b=*/false);
}

// Branch-free tanh on the serial critical path: prefer hw v_tanh_f32,
// else tanh(x) = 1 - 2*rcp(exp2(2*log2e*x) + 1)  (saturates to +-1, no branches).
static __device__ __forceinline__ float fast_tanh(float v) {
#if __has_builtin(__builtin_amdgcn_tanhf)
    return __builtin_amdgcn_tanhf(v);
#else
    float e = __builtin_amdgcn_exp2f(v * 2.8853900817779268f); // 2*log2(e)
    return 1.0f - 2.0f * __builtin_amdgcn_rcpf(e + 1.0f);
#endif
}

static __device__ __forceinline__ float fast_sigmoid(float z) {
    return __builtin_amdgcn_rcpf(1.0f + __builtin_amdgcn_exp2f(-z * 1.4426950408889634f));
}

// ---------------------------------------------------------------------------
// Kernel 0: convert fc1_W (f32, 64 x 131072) -> f16 in workspace
// ---------------------------------------------------------------------------
__global__ __launch_bounds__(256) void cvt_fc1_kernel(
    const float* __restrict__ src, _Float16* __restrict__ dst, int nelem)
{
    int i = (blockIdx.x * 256 + threadIdx.x) * 4;
    if (i + 3 < nelem) {
        float4 f = *(const float4*)(src + i);
        v4h h;
        h[0] = (_Float16)f.x; h[1] = (_Float16)f.y;
        h[2] = (_Float16)f.z; h[3] = (_Float16)f.w;
        *(v4h*)(dst + i) = h;
    }
}

// ---------------------------------------------------------------------------
// Kernel 1: RNN recurrence. 16 workgroups (one per 16-row batch tile),
// 256 threads = 8 waves; wave w owns H columns [16w, 16w+16).
// Per step: 8 ds_load_b128 -> 4 chained v_wmma_f32_16x16x32_f16 -> tanh ->
// LDS + f16 global store -> one barrier. W_hh B-frags live in VGPRs; the
// whole 16x1024 x-tile lives in LDS (loaded once).
// ---------------------------------------------------------------------------
__global__ __launch_bounds__(256) void rnn_kernel(
    const float* __restrict__ x,     // (B, 1, L)
    const float* __restrict__ W_ih,  // (H, 1)
    const float* __restrict__ b_ih,  // (H,)
    const float* __restrict__ W_hh,  // (H, H)
    const float* __restrict__ b_hh,  // (H,)
    _Float16*    __restrict__ flat)  // (B, L*H) f16 out
{
    __shared__ _Float16 hbuf[2][16][H_];  // ping-pong hidden state tile (8 KB)
    __shared__ float    xtile[16][L_];    // all x for this batch tile (64 KB)

    const int tid  = threadIdx.x;
    const int lane = tid & 31;
    const int wv   = tid >> 5;            // 0..7
    const int b0   = blockIdx.x * 16;     // batch tile base
    const int lo16 = lane & 15;
    const int hi16 = lane >> 4;           // 0 or 1
    const int n    = wv * 16 + lo16;      // this lane's output column (H index)

    // Preload B fragments of W_hh^T: B[k,n] = W_hh[n,k].
    // 16-bit B 32x16 layout: lanes 0-15: K {0..7,16..23}; lanes 16-31: K {8..15,24..31}.
    v16h bw[4];
    #pragma unroll
    for (int c = 0; c < 4; ++c) {
        const float* wp = W_hh + (size_t)n * H_ + c * 32 + (hi16 ? 8 : 0);
        #pragma unroll
        for (int i = 0; i < 16; ++i) bw[c][i] = (_Float16)wp[i < 8 ? i : i + 8];
    }
    const float wih  = W_ih[n];
    const float bias = b_ih[n] + b_hh[n];

    // Load full x tile (16 rows x 1024) into LDS, coalesced float4s.
    {
        const float4* xsrc = (const float4*)(x + (size_t)b0 * L_);
        float4*       xdst = (float4*)&xtile[0][0];
        #pragma unroll
        for (int i = 0; i < (16 * L_ / 4) / 256; ++i)
            xdst[tid + i * 256] = xsrc[tid + i * 256];
    }
    // h0 = 0.
    {
        v8h z = {};
        ((v8h*)&hbuf[0][0][0])[tid] = z;   // 2048 halves / 256 threads
    }
    __syncthreads();

    for (int t = 0; t < L_; ++t) {
        const int cur = t & 1, nxt = cur ^ 1;

        // acc = h(16x128) @ W_hh^T tile (128x16), K in 4 chunks of 32.
        v8f acc = {};
        #pragma unroll
        for (int c = 0; c < 4; ++c) {
            // 16-bit A 16x32 layout: lanes 0-15: K {0..7,16..23}; lanes 16-31: K {8..15,24..31}
            const _Float16* rp = &hbuf[cur][lo16][c * 32 + (hi16 ? 8 : 0)];
            v8h alo = *(const v8h*)rp;         // ds_load_b128
            v8h ahi = *(const v8h*)(rp + 16);  // ds_load_b128
            v16h a;
            #pragma unroll
            for (int i = 0; i < 8; ++i) { a[i] = alo[i]; a[8 + i] = ahi[i]; }
            acc = wmma_f16(a, bw[c], acc);
        }

        // epilogue: tanh(acc + x*W_ih + b_ih + b_hh); write next h + flat.
        #pragma unroll
        for (int v = 0; v < 8; ++v) {
            const int m = hi16 * 8 + v;       // C layout: M = 8*hi16 + vgpr
            float hn = fast_tanh(__builtin_fmaf(xtile[m][t], wih, bias) + acc[v]);
            _Float16 hh = (_Float16)hn;
            hbuf[nxt][m][n] = hh;
            flat[(size_t)(b0 + m) * K_TOT + (size_t)t * H_ + n] = hh;
        }
        __syncthreads();
    }
}

// ---------------------------------------------------------------------------
// Kernel 2: h1 = relu(flat @ fc1_W^T + fc1_b) fused with
//           out = sigmoid(h1 @ fc2_W^T + fc2_b).
// 16 workgroups x 128 threads (4 waves); wave w owns FC1 columns [16w,16w+16).
// ---------------------------------------------------------------------------
__global__ __launch_bounds__(128) void fc_kernel(
    const _Float16* __restrict__ flat,   // (B, 131072) f16
    const _Float16* __restrict__ fc1h,   // (64, 131072) f16
    const float*    __restrict__ fc1_b,  // (64,)
    const float*    __restrict__ fc2_W,  // (2, 64)
    const float*    __restrict__ fc2_b,  // (2,)
    float*          __restrict__ out)    // (B, 2)
{
    __shared__ float h1s[16][FC1_];

    const int tid  = threadIdx.x;
    const int lane = tid & 31;
    const int wv   = tid >> 5;          // 0..3
    const int m0   = blockIdx.x * 16;   // batch tile base
    const int lo16 = lane & 15;
    const int hi16 = lane >> 4;
    const int n    = wv * 16 + lo16;    // FC1 column

    const _Float16* ap = flat + (size_t)(m0 + lo16) * K_TOT + (hi16 ? 8 : 0);
    const _Float16* bp = fc1h + (size_t)n * K_TOT + (hi16 ? 8 : 0);

    v8f acc = {};
    #pragma unroll 2
    for (int c = 0; c < K_TOT / 32; ++c) {
        v8h alo = *(const v8h*)(ap + c * 32);       // global_load_b128
        v8h ahi = *(const v8h*)(ap + c * 32 + 16);
        v8h blo = *(const v8h*)(bp + c * 32);
        v8h bhi = *(const v8h*)(bp + c * 32 + 16);
        v16h a, b;
        #pragma unroll
        for (int i = 0; i < 8; ++i) {
            a[i] = alo[i]; a[8 + i] = ahi[i];
            b[i] = blo[i]; b[8 + i] = bhi[i];
        }
        acc = wmma_f16(a, b, acc);
    }

    const float fb = fc1_b[n];
    #pragma unroll
    for (int v = 0; v < 8; ++v) {
        const int m = hi16 * 8 + v;
        float h1 = acc[v] + fb;
        h1s[m][n] = h1 > 0.f ? h1 : 0.f;
    }
    __syncthreads();

    if (tid < 16) {
        float o0 = fc2_b[0], o1 = fc2_b[1];
        #pragma unroll 8
        for (int j = 0; j < FC1_; ++j) {
            float hv = h1s[tid][j];
            o0 = __builtin_fmaf(hv, fc2_W[j], o0);
            o1 = __builtin_fmaf(hv, fc2_W[FC1_ + j], o1);
        }
        out[(size_t)(m0 + tid) * 2 + 0] = fast_sigmoid(o0);
        out[(size_t)(m0 + tid) * 2 + 1] = fast_sigmoid(o1);
    }
}

// ---------------------------------------------------------------------------
extern "C" void kernel_launch(void* const* d_in, const int* in_sizes, int n_in,
                              void* d_out, int out_size, void* d_ws, size_t ws_size,
                              hipStream_t stream)
{
    const float* x     = (const float*)d_in[0];  // (256,1,1024)
    const float* W_ih  = (const float*)d_in[1];  // (128,1)
    const float* b_ih  = (const float*)d_in[2];  // (128,)
    const float* W_hh  = (const float*)d_in[3];  // (128,128)
    const float* b_hh  = (const float*)d_in[4];  // (128,)
    const float* fc1_W = (const float*)d_in[5];  // (64,131072)
    const float* fc1_b = (const float*)d_in[6];  // (64,)
    const float* fc2_W = (const float*)d_in[7];  // (2,64)
    const float* fc2_b = (const float*)d_in[8];  // (2,)
    float* out = (float*)d_out;                  // (256,2) f32

    // Workspace: [0, 64MiB) flat f16 activations; then fc1_W in f16 (16 MiB).
    _Float16* flat = (_Float16*)d_ws;
    _Float16* fc1h = (_Float16*)((char*)d_ws + (size_t)B_ * K_TOT * sizeof(_Float16));

    const int ncvt = FC1_ * K_TOT;                       // 8388608
    cvt_fc1_kernel<<<ncvt / (256 * 4), 256, 0, stream>>>(fc1_W, fc1h, ncvt);
    rnn_kernel<<<B_ / 16, 256, 0, stream>>>(x, W_ih, b_ih, W_hh, b_hh, flat);
    fc_kernel<<<B_ / 16, 128, 0, stream>>>(flat, fc1h, fc1_b, fc2_W, fc2_b, out);
}
